// Model_52115133170028
// MI455X (gfx1250) — matrix operand order repaired and checked by reading, not compiled, not run on hardware
//
#include <hip/hip_runtime.h>

// ---------------------------------------------------------------------------
// Deep BSDE (100-d HJB) solver for MI455X (gfx1250, wave32, WMMA bf16).
// ---------------------------------------------------------------------------

typedef __attribute__((ext_vector_type(16))) __bf16 v16bf;
typedef __attribute__((ext_vector_type(8)))  __bf16 v8bf;
typedef __attribute__((ext_vector_type(8)))  float  v8f;

// Problem constants
#define BATCH   16384
#define NSTEPS  64
#define DH      512     // hidden
#define DX      100     // x / z dim
#define KU      128     // padded input dim (100 x + 1 t + 27 zero)
#define MT      64      // batch rows per workgroup
#define XS      104     // f32 row stride for x/w buffers
#define US      136     // bf16 row stride for u buffer  (272B -> 4-bank skew)
#define HS      520     // bf16 row stride for h buffers (1040B -> 4-bank skew)

#define SMEM_BYTES 204544

// Native f32 -> bf16 conversion (hardware RNE convert on gfx1250).
__device__ __forceinline__ __bf16 f2bf(float f) {
    return static_cast<__bf16>(f);
}

union Frag { v16bf v; v8bf h[2]; };

// A fragment (16x32 bf16) from a row-major bf16 matrix (LDS).
// Per ISA 7.12.2: lane m=lane&15, elems 0..7 = K[base..base+7],
// elems 8..15 = K[base+16..base+23], base = kc*32 + (lane>>4)*8.
__device__ __forceinline__ v16bf load_afrag(const __bf16* mat, int stride,
                                            int lane, int kc) {
    const int m = lane & 15;
    const int k = kc * 32 + ((lane >> 4) << 3);
    const __bf16* p = mat + m * stride + k;
    Frag f;
    f.h[0] = *(const v8bf*)(p);
    f.h[1] = *(const v8bf*)(p + 16);
    return f.v;
}

// B fragment (32x16 bf16) from prepacked weights: per-lane 32B contiguous.
__device__ __forceinline__ v16bf load_bfrag(const __bf16* pack, int Nc,
                                            int kc, int nt, int lane) {
    const __bf16* p = pack + ((((size_t)(kc * Nc + nt)) * 32 + lane) << 4);
    Frag f;
    f.h[0] = ((const v8bf*)p)[0];
    f.h[1] = ((const v8bf*)p)[1];
    return f.v;
}

__device__ __forceinline__ v8f wmma_bf16(v16bf a, v16bf b, v8f c) {
    return __builtin_amdgcn_wmma_f32_16x16x32_bf16(
        /*neg_a=*/false, a, /*neg_b=*/false, b,
        /*c_mod=*/(short)0, c, /*reuse_a=*/false, /*reuse_b=*/false);
}

// ---------------------------------------------------------------------------
// Prepass 1: pack W[n][k] (row-major [Nreal][Kreal]) into bf16 B-fragment
// layout for W^T: pack[((kc*Nc+nt)*32+lane)*16 + e] = W[n][k],
//   k = kc*32 + (lane>>4)*8 + (e&7) + (e>>3)*16,  n = nt*16 + (lane&15), zero padded.
// ---------------------------------------------------------------------------
__global__ void pack_weights_kernel(const float* __restrict__ W,
                                    __bf16* __restrict__ dst,
                                    int Kreal, int Nreal, int Kc, int Nc) {
    const int total = Kc * Nc * 512;
    for (int idx = blockIdx.x * blockDim.x + threadIdx.x; idx < total;
         idx += gridDim.x * blockDim.x) {
        const int e    = idx & 15;
        const int lane = (idx >> 4) & 31;
        const int fr   = idx >> 9;
        const int nt   = fr % Nc;
        const int kc   = fr / Nc;
        const int k = kc * 32 + ((lane >> 4) << 3) + (e & 7) + ((e >> 3) << 4);
        const int n = nt * 16 + (lane & 15);
        const float v = (k < Kreal && n < Nreal) ? W[(size_t)n * Kreal + k] : 0.f;
        dst[idx] = f2bf(v);
    }
}

// ---------------------------------------------------------------------------
// Prepass 2: transpose dW [B*DX][NSTEPS] -> dWt [NSTEPS][B*DX] (coalesced
// per-step reads in main kernel).  R = BATCH*DX is a multiple of 32.
// ---------------------------------------------------------------------------
__global__ void transpose_dw_kernel(const float* __restrict__ in,
                                    float* __restrict__ out) {
    __shared__ float tile[32][33];
    const size_t R = (size_t)BATCH * DX;
    const int rb = blockIdx.x * 32;
    const int sb = blockIdx.y * 32;
    const int tx = threadIdx.x, ty = threadIdx.y;   // (32, 8)
    #pragma unroll
    for (int yy = 0; yy < 32; yy += 8)
        tile[ty + yy][tx] = in[(size_t)(rb + ty + yy) * NSTEPS + sb + tx];
    __syncthreads();
    #pragma unroll
    for (int yy = 0; yy < 32; yy += 8)
        out[(size_t)(sb + ty + yy) * R + rb + tx] = tile[tx][ty + yy];
}

// ---------------------------------------------------------------------------
// Main kernel: 256 threads (8 wave32s) per WG, 64 batch rows per WG.
// Wave w: M-tile mt = w&3 (16 rows), N-half nh = w>>2.
// ---------------------------------------------------------------------------
__global__ void __launch_bounds__(256, 1)
bsde_kernel(const float* __restrict__ b1, const float* __restrict__ b2,
            const float* __restrict__ b3, const float* __restrict__ y0,
            const __bf16* __restrict__ pW1, const __bf16* __restrict__ pW2,
            const __bf16* __restrict__ pW3, const float* __restrict__ dWt,
            float* __restrict__ out) {
    extern __shared__ char smem[];
    float*  xbuf  = (float*)smem;                  // [MT][XS] f32
    float*  wbuf  = xbuf + MT * XS;                // [MT][XS] f32
    float*  ybuf  = wbuf + MT * XS;                // [MT]     f32
    __bf16* ubuf  = (__bf16*)(ybuf + MT);          // [MT][US] bf16
    __bf16* h1buf = ubuf + MT * US;                // [MT][HS] bf16
    __bf16* h2buf = h1buf + MT * HS;               // [MT][HS] bf16
    float*  ypart = (float*)(h2buf + MT * HS);     // [8][16]  f32

    const int tid  = threadIdx.x;
    const int lane = tid & 31;
    const int w    = tid >> 5;
    const int mt   = w & 3;
    const int nh   = w >> 2;
    const int g0   = blockIdx.x * MT;

    const float dt    = 1.0f / 64.0f;
    const float sqdt  = 0.125f;                    // sqrt(dt)
    const float sq2   = 1.41421356237309515f;      // sqrt(2)
    const float lamdt = dt;                        // LAM * dt, LAM = 1

    // ---- init state ----
    const float y0v = y0[0];
    for (int i = tid; i < MT * XS; i += 256) xbuf[i] = 0.f;   // X0 = 0
    if (tid < MT) ybuf[tid] = y0v;
    __syncthreads();

    for (int step = 0; step < NSTEPS; ++step) {
        const float t = step * dt;

        // ---- stage w slice (coalesced), build u = [x, t, 0...] in bf16 ----
        if (tid < 128) ypart[tid] = 0.f;
        const float* dws = dWt + ((size_t)step * BATCH + g0) * DX;
        for (int i = tid; i < MT * DX; i += 256) {
            const int r = i / DX, d = i - r * DX;
            wbuf[r * XS + d] = sqdt * dws[i];
        }
        for (int i = tid; i < MT * KU; i += 256) {
            const int r = i >> 7, k = i & 127;
            const float v = (k < DX) ? xbuf[r * XS + k] : (k == DX ? t : 0.f);
            ubuf[r * US + k] = f2bf(v);
        }
        __syncthreads();

        // ---- x update (z below uses the pre-update x snapshot in ubuf) ----
        for (int i = tid; i < MT * DX; i += 256) {
            const int r = i / DX, d = i - r * DX;
            xbuf[r * XS + d] += sq2 * wbuf[r * XS + d];
        }

        // ---- layer 1: h1 = relu(u @ W1^T + b1), K=128, N=512 ----
        {
            const __bf16* arow = ubuf + mt * 16 * US;
            const v16bf a0 = load_afrag(arow, US, lane, 0);
            const v16bf a1 = load_afrag(arow, US, lane, 1);
            const v16bf a2 = load_afrag(arow, US, lane, 2);
            const v16bf a3 = load_afrag(arow, US, lane, 3);
            for (int nt = nh * 16; nt < nh * 16 + 16; ++nt) {
                const float bias = b1[nt * 16 + (lane & 15)];
                v8f c;
                #pragma unroll
                for (int j = 0; j < 8; ++j) c[j] = bias;
                c = wmma_bf16(a0, load_bfrag(pW1, 32, 0, nt, lane), c);
                c = wmma_bf16(a1, load_bfrag(pW1, 32, 1, nt, lane), c);
                c = wmma_bf16(a2, load_bfrag(pW1, 32, 2, nt, lane), c);
                c = wmma_bf16(a3, load_bfrag(pW1, 32, 3, nt, lane), c);
                const int mrow = mt * 16 + ((lane >> 4) << 3);
                const int ncol = nt * 16 + (lane & 15);
                #pragma unroll
                for (int j = 0; j < 8; ++j)
                    h1buf[(mrow + j) * HS + ncol] = f2bf(fmaxf(c[j], 0.f));
            }
        }
        __syncthreads();

        // ---- layer 2: h2 = relu(h1 @ W2^T + b2), K=512, N=512 ----
        {
            const __bf16* arow = h1buf + mt * 16 * HS;
            v16bf a[16];
            #pragma unroll
            for (int kc = 0; kc < 16; ++kc) a[kc] = load_afrag(arow, HS, lane, kc);
            for (int nt = nh * 16; nt < nh * 16 + 16; ++nt) {
                const float bias = b2[nt * 16 + (lane & 15)];
                v8f c;
                #pragma unroll
                for (int j = 0; j < 8; ++j) c[j] = bias;
                #pragma unroll
                for (int kc = 0; kc < 16; ++kc)
                    c = wmma_bf16(a[kc], load_bfrag(pW2, 32, kc, nt, lane), c);
                const int mrow = mt * 16 + ((lane >> 4) << 3);
                const int ncol = nt * 16 + (lane & 15);
                #pragma unroll
                for (int j = 0; j < 8; ++j)
                    h2buf[(mrow + j) * HS + ncol] = f2bf(fmaxf(c[j], 0.f));
            }
        }
        __syncthreads();

        // ---- layer 3: z = h2 @ W3^T + b3 (N padded 112), then y update ----
        {
            const __bf16* arow = h2buf + mt * 16 * HS;
            v16bf a[16];
            #pragma unroll
            for (int kc = 0; kc < 16; ++kc) a[kc] = load_afrag(arow, HS, lane, kc);
            for (int q = 0; q < 4; ++q) {
                const int nt = nh + 2 * q;        // nh=0: 0,2,4,6  nh=1: 1,3,5
                if (nt >= 7) break;               // wave-uniform
                const int d = nt * 16 + (lane & 15);
                const float bias = (d < DX) ? b3[d] : 0.f;
                v8f c;
                #pragma unroll
                for (int j = 0; j < 8; ++j) c[j] = bias;
                #pragma unroll
                for (int kc = 0; kc < 16; ++kc)
                    c = wmma_bf16(a[kc], load_bfrag(pW3, 7, kc, nt, lane), c);
                const int mr = (lane >> 4) << 3;  // local row base (0 or 8)
                #pragma unroll
                for (int j = 0; j < 8; ++j) {
                    const int rl = mt * 16 + mr + j;
                    float contrib = 0.f;
                    if (d < DX) {
                        const float z  = c[j];
                        const float wv = wbuf[rl * XS + d];
                        contrib = z * (lamdt * z + wv);  // lam*dt*z^2 + z*w
                    }
                    // reduce over the 16 lanes sharing this row
                    contrib += __shfl_xor(contrib, 1, 32);
                    contrib += __shfl_xor(contrib, 2, 32);
                    contrib += __shfl_xor(contrib, 4, 32);
                    contrib += __shfl_xor(contrib, 8, 32);
                    if ((lane & 15) == 0) ypart[w * 16 + mr + j] += contrib;
                }
            }
        }
        __syncthreads();

        // ---- deterministic y accumulation (fixed order) ----
        if (tid < MT) {
            const int m = tid >> 4, r = tid & 15;
            ybuf[tid] += ypart[m * 16 + r] + ypart[(m + 4) * 16 + r];
        }
        __syncthreads();
    }

    // ---- outputs: x [BATCH][DX], then y [BATCH] ----
    for (int i = tid; i < MT * DX; i += 256) {
        const int r = i / DX, d = i - r * DX;
        out[(size_t)(g0 + r) * DX + d] = xbuf[r * XS + d];
    }
    if (tid < MT) out[(size_t)BATCH * DX + g0 + tid] = ybuf[tid];
}

// ---------------------------------------------------------------------------
// Launcher
// ---------------------------------------------------------------------------
extern "C" void kernel_launch(void* const* d_in, const int* in_sizes, int n_in,
                              void* d_out, int out_size, void* d_ws,
                              size_t ws_size, hipStream_t stream) {
    const float* W1 = (const float*)d_in[0];   // [512][101]
    const float* b1 = (const float*)d_in[1];   // [512]
    const float* W2 = (const float*)d_in[2];   // [512][512]
    const float* b2 = (const float*)d_in[3];   // [512]
    const float* W3 = (const float*)d_in[4];   // [100][512]
    const float* b3 = (const float*)d_in[5];   // [100]
    const float* y0 = (const float*)d_in[6];   // [1]
    const float* dW = (const float*)d_in[7];   // [16384][100][64]
    float* out = (float*)d_out;

    char* ws = (char*)d_ws;
    float* dWt = (float*)ws;                                   // [64][B][100]
    const size_t dwt_bytes = (size_t)NSTEPS * BATCH * DX * sizeof(float);
    __bf16* pW1 = (__bf16*)(ws + dwt_bytes);   //  4*32*512 elems
    __bf16* pW2 = pW1 + 4 * 32 * 512;          // 16*32*512 elems
    __bf16* pW3 = pW2 + 16 * 32 * 512;         // 16* 7*512 elems

    pack_weights_kernel<<<256, 256, 0, stream>>>(W1, pW1, 101, 512, 4, 32);
    pack_weights_kernel<<<1024, 256, 0, stream>>>(W2, pW2, 512, 512, 16, 32);
    pack_weights_kernel<<<256, 256, 0, stream>>>(W3, pW3, 512, 100, 16, 7);

    dim3 tgrid((BATCH * DX) / 32, NSTEPS / 32);
    transpose_dw_kernel<<<tgrid, dim3(32, 8), 0, stream>>>(dW, dWt);

    hipFuncSetAttribute(reinterpret_cast<const void*>(bsde_kernel),
                        hipFuncAttributeMaxDynamicSharedMemorySize, SMEM_BYTES);
    bsde_kernel<<<BATCH / MT, 256, SMEM_BYTES, stream>>>(
        b1, b2, b3, y0, pW1, pW2, pW3, dWt, out);
}
